// LSTMmodel_10170482557259
// MI455X (gfx1250) — matrix operand order repaired and checked by reading, not compiled, not run on hardware
//
#include <hip/hip_runtime.h>
#include <math.h>

// ---------------- model dims ----------------
#define B_    32
#define LPRE  128
#define LHYP  128
#define H_    1024
#define E_    512
#define G4H   4096     // 4*H
#define NCLS_ 3

typedef _Float16 h16;
typedef __attribute__((ext_vector_type(16))) _Float16 v16h;
typedef __attribute__((ext_vector_type(8)))  _Float16 v8h;
typedef __attribute__((ext_vector_type(8)))  float    v8f;

// ---------------- WMMA helpers ----------------
// D = A(16x32 f16) * B(32x16 f16) + C(16x16 f32), wave32
__device__ __forceinline__ v8f wmma16(v16h a, v16h b, v8f c) {
  return __builtin_amdgcn_wmma_f32_16x16x32_f16(
      /*neg_a=*/false, a, /*neg_b=*/false, b,
      /*c_mod=*/(short)0, c, /*reuse_a=*/false, /*reuse_b=*/false);
}

// A-matrix 16x32 f16 tile from row-major [rows][ldk] source (global or LDS).
// Per ISA: lanes 0-15 hold M=0..15; g=lane>>4 selects K-half:
//   elems 0..7  -> K = k0 + g*8 + {0..7}
//   elems 8..15 -> K = k0 + 16 + g*8 + {0..7}
__device__ __forceinline__ v16h load_a_f16(const h16* __restrict__ base, int ldk,
                                           int row0, int k0, int lane) {
  int m = row0 + (lane & 15);
  int g = lane >> 4;
  const h16* p = base + (size_t)m * ldk + k0 + g * 8;
  v8h lo = *(const v8h*)(p);
  v8h hi = *(const v8h*)(p + 16);
  v16h r;
#pragma unroll
  for (int i = 0; i < 8; ++i) { r[i] = lo[i]; r[i + 8] = hi[i]; }
  return r;
}

// Same A layout but converting f32 source on the fly.
__device__ __forceinline__ v16h load_a_f32(const float* __restrict__ base, int ldk,
                                           int row0, int k0, int lane) {
  int m = row0 + (lane & 15);
  int g = lane >> 4;
  const float* p = base + (size_t)m * ldk + k0 + g * 8;
  v16h r;
#pragma unroll
  for (int i = 0; i < 8; ++i) { r[i] = (h16)p[i]; r[i + 8] = (h16)p[16 + i]; }
  return r;
}

// B-matrix 32x16 f16 tile, B[k][n] sourced from row-major W[n][k] ([cols][ldk]).
// Per ISA (dense 16-bit B): lane holds N=lane&15; g=lane>>4 selects K-half:
//   elems 0..7  -> K = k0 + g*8 + {0..7}
//   elems 8..15 -> K = k0 + 16 + g*8 + {0..7}
__device__ __forceinline__ v16h load_b_f16(const h16* __restrict__ base, int ldk,
                                           int col0, int k0, int lane) {
  int n = col0 + (lane & 15);
  int g = lane >> 4;
  const h16* p = base + (size_t)n * ldk + k0 + g * 8;
  v8h lo = *(const v8h*)(p);
  v8h hi = *(const v8h*)(p + 16);
  v16h r;
#pragma unroll
  for (int i = 0; i < 8; ++i) { r[i] = lo[i]; r[i + 8] = hi[i]; }
  return r;
}

// Async global -> LDS copy, 16 bytes per lane (CDNA5 GLOBAL_LOAD_ASYNC_TO_LDS_B128,
// tracked by ASYNCcnt). LDS address = low 32 bits of generic pointer (aperture rule).
__device__ __forceinline__ void async_cp16(h16* lds_dst, const h16* gsrc) {
  unsigned ldsoff = (unsigned)(unsigned long long)lds_dst;
  asm volatile("global_load_async_to_lds_b128 %0, %1, off"
               :
               : "v"(ldsoff), "v"(gsrc)
               : "memory");
}
__device__ __forceinline__ void wait_async0() {
  asm volatile("s_wait_asynccnt 0" ::: "memory");
}

__device__ __forceinline__ float sigf(float x) { return 1.f / (1.f + __expf(-x)); }

// ---------------- prep: casts, embeddings, zero-init ----------------
__global__ void prep_kernel(const float* __restrict__ emb,
                            const int* __restrict__ Xpre, const int* __restrict__ Xhyp,
                            const float* __restrict__ WihA, const float* __restrict__ WhhA,
                            const float* __restrict__ bihA, const float* __restrict__ bhhA,
                            const float* __restrict__ WihB, const float* __restrict__ WhhB,
                            const float* __restrict__ bihB, const float* __restrict__ bhhB,
                            h16* __restrict__ wihA16, h16* __restrict__ whhA16,
                            h16* __restrict__ wihB16, h16* __restrict__ whhB16,
                            h16* __restrict__ xp, h16* __restrict__ xh,
                            float* __restrict__ biasA, float* __restrict__ biasB,
                            h16* __restrict__ hbuf, float* __restrict__ ascore,
                            int* __restrict__ ctr) {
  size_t tid = (size_t)blockIdx.x * blockDim.x + threadIdx.x;
  size_t stride = (size_t)gridDim.x * blockDim.x;
  const size_t nIh = (size_t)G4H * E_;
  const size_t nHh = (size_t)G4H * H_;
  for (size_t i = tid; i < nIh; i += stride) { wihA16[i] = (h16)WihA[i]; wihB16[i] = (h16)WihB[i]; }
  for (size_t i = tid; i < nHh; i += stride) { whhA16[i] = (h16)WhhA[i]; whhB16[i] = (h16)WhhB[i]; }
  for (size_t i = tid; i < (size_t)G4H; i += stride) {
    biasA[i] = bihA[i] + bhhA[i];
    biasB[i] = bihB[i] + bhhB[i];
  }
  // xp[l][b][e] = emb[Xpre[b][l]][e]
  const size_t nEmb = (size_t)LPRE * B_ * E_;
  for (size_t i = tid; i < nEmb; i += stride) {
    size_t e = i % E_;
    size_t lb = i / E_;
    size_t b = lb % B_;
    size_t l = lb / B_;
    xp[i] = (h16)emb[(size_t)Xpre[b * LPRE + l] * E_ + e];
    xh[i] = (h16)emb[(size_t)Xhyp[b * LHYP + l] * E_ + e];
  }
  for (size_t i = tid; i < (size_t)B_ * H_; i += stride) hbuf[i] = (h16)0.f;       // h0 = 0 (buffer 0)
  for (size_t i = tid; i < (size_t)B_ * LPRE; i += stride) ascore[i] = 0.f;         // attention accum
  if (tid == 0) *ctr = 0;                                                            // grid barrier counter
}

// ---------------- persistent LSTM recurrence (256 sequential steps) ----------------
// 32 workgroups x 256 threads (8 waves). WG `wg` owns hidden units [wg*32, wg*32+32).
// Gates G(32x4096) = [x_t|h] (32x1536) * [Wih|Whh]^T via f16 WMMA, K fused 512+1024.
// x_t (32 KB) and h (64 KB) are staged into LDS with async global->LDS loads each
// step and shared by all 8 waves. c kept in LDS across all 256 steps; h double-
// buffered in global; device-wide barrier per step.
//
// Dynamic LDS layout (116 KB total):
//   [0,       32 KB)  xS      : x_t   f16 [B][E]
//   [32 KB,   96 KB)  hS      : h_in  f16 [B][H]
//   [96 KB,  112 KB)  gateBuf : f32 [B][128]
//   [112 KB, 116 KB)  cSl     : f32 [B][32]
#define LDS_BYTES (116 * 1024)
__global__ __launch_bounds__(256) void lstm_persistent(
    const h16* __restrict__ xp, const h16* __restrict__ xh,
    const h16* __restrict__ wihA, const h16* __restrict__ whhA,
    const h16* __restrict__ wihB, const h16* __restrict__ whhB,
    const float* __restrict__ biasA, const float* __restrict__ biasB,
    h16* __restrict__ hbuf,     // [2][B][H] f16, double-buffered
    float* __restrict__ hf32,   // [B][H] f32 (final = h_last)
    h16* __restrict__ hspre,    // [LPRE][B][H] f16 premise hidden states
    int* __restrict__ ctr) {
  extern __shared__ char smem[];
  h16*   xS      = (h16*)(smem);
  h16*   hS      = (h16*)(smem + 32 * 1024);
  float* gateBuf = (float*)(smem + 96 * 1024);
  float* cSl     = (float*)(smem + 112 * 1024);

  const int tid = threadIdx.x;
  const int lane = tid & 31;
  const int w = tid >> 5;       // wave 0..7
  const int wg = blockIdx.x;    // 0..31

  for (int i = tid; i < B_ * 32; i += 256) cSl[i] = 0.f;  // c0 = 0

  // wave -> tile mapping: t = mi*8 + gi*2 + nc ; wave w owns t=2w,2w+1 (shared A tile)
  const int mi = w >> 2;            // M-tile 0..1 (batch rows m0..m0+15)
  const int gi = w & 3;             // gate index i/f/g/o
  const int m0 = mi * 16;
  const int nG0 = gi * H_ + wg * 32;  // global gate-column base (nc=0)
  const int nL0 = gi * 32;            // local column base in gateBuf
  __syncthreads();

  const int nSteps = LPRE + LHYP;
  for (int s = 0; s < nSteps; ++s) {
    const int seq = s >> 7;
    const int l = s & 127;
    const h16* Xc  = (seq ? xh : xp) + (size_t)l * B_ * E_;
    const h16* Wih = seq ? wihB : wihA;
    const h16* Whh = seq ? whhB : whhA;
    const float* bias = seq ? biasB : biasA;
    const h16* hin = hbuf + (size_t)(s & 1) * B_ * H_;
    h16* hout      = hbuf + (size_t)((s + 1) & 1) * B_ * H_;

    // ---- stage x_t and h into LDS via async global->LDS (ASYNCcnt) ----
#pragma unroll
    for (int i = 0; i < (B_ * E_ / 8) / 256; ++i) {          // 8 chunks/thread
      int c16 = tid + i * 256;
      async_cp16(xS + c16 * 8, Xc + c16 * 8);
    }
#pragma unroll
    for (int i = 0; i < (B_ * H_ / 8) / 256; ++i) {          // 16 chunks/thread
      int c16 = tid + i * 256;
      async_cp16(hS + c16 * 8, hin + c16 * 8);
    }
    wait_async0();
    __syncthreads();

    v8f acc0 = {};
    v8f acc1 = {};
    // input projection part: K = 512 (A from LDS, B = weights from global/L2)
#pragma unroll 2
    for (int k = 0; k < E_; k += 32) {
      v16h a  = load_a_f16(xS, E_, m0, k, lane);
      v16h b0 = load_b_f16(Wih, E_, nG0,      k, lane);
      v16h b1 = load_b_f16(Wih, E_, nG0 + 16, k, lane);
      acc0 = wmma16(a, b0, acc0);
      acc1 = wmma16(a, b1, acc1);
    }
    // recurrent part: K = 1024
#pragma unroll 2
    for (int k = 0; k < H_; k += 32) {
      v16h a  = load_a_f16(hS, H_, m0, k, lane);
      v16h b0 = load_b_f16(Whh, H_, nG0,      k, lane);
      v16h b1 = load_b_f16(Whh, H_, nG0 + 16, k, lane);
      acc0 = wmma16(a, b0, acc0);
      acc1 = wmma16(a, b1, acc1);
    }
    // D layout: lane n=lane&15 holds column; VGPR r holds row m = r + 8*(lane>>4)
    {
      const int n = lane & 15;
      const int g = lane >> 4;
#pragma unroll
      for (int r0 = 0; r0 < 8; ++r0) {
        int m = m0 + r0 + 8 * g;
        gateBuf[m * 128 + nL0 + n]      = acc0[r0];
        gateBuf[m * 128 + nL0 + 16 + n] = acc1[r0];
      }
    }
    __syncthreads();

    // elementwise LSTM cell for this WG's 32 units x 32 batches
#pragma unroll
    for (int p = 0; p < 4; ++p) {
      int idx = tid + p * 256;           // 0..1023
      int b = idx >> 5;
      int u = idx & 31;
      int j = wg * 32 + u;
      float iv = sigf(gateBuf[b * 128 + u]      + bias[j]);
      float fv = sigf(gateBuf[b * 128 + 32 + u] + bias[H_ + j]);
      float gv = tanhf(gateBuf[b * 128 + 64 + u] + bias[2 * H_ + j]);
      float ov = sigf(gateBuf[b * 128 + 96 + u] + bias[3 * H_ + j]);
      float c = fv * cSl[idx] + iv * gv;
      cSl[idx] = c;
      float h = ov * tanhf(c);
      hout[(size_t)b * H_ + j] = (h16)h;
      hf32[(size_t)b * H_ + j] = h;
      if (seq == 0) hspre[((size_t)l * B_ + b) * H_ + j] = (h16)h;
    }

    // device-wide barrier (monotone counter; 32 WGs guaranteed co-resident)
    __syncthreads();
    if (tid == 0) {
      __threadfence();
      atomicAdd(ctr, 1);
      const int target = 32 * (s + 1);
      while (__hip_atomic_load(ctr, __ATOMIC_ACQUIRE, __HIP_MEMORY_SCOPE_AGENT) < target) {
        __builtin_amdgcn_s_sleep(1);
      }
    }
    __syncthreads();
  }
}

// ---------------- batched GEMV: out[b][j] = sum_h W[b][j][h] * v[b][h] ----------------
__global__ __launch_bounds__(256) void gemv_bhh(const float* __restrict__ W,
                                                const float* __restrict__ v,
                                                float* __restrict__ out, int accumulate) {
  int row = blockIdx.x * 8 + (threadIdx.x >> 5);  // 0..32767
  int lane = threadIdx.x & 31;
  int b = row >> 10;
  int j = row & 1023;
  const float* wr = W + (size_t)b * H_ * H_ + (size_t)j * H_;
  const float* vr = v + (size_t)b * H_;
  float s = 0.f;
  for (int h = lane; h < H_; h += 32) s += wr[h] * vr[h];
#pragma unroll
  for (int off = 16; off; off >>= 1) s += __shfl_down(s, off, 32);
  if (lane == 0) {
    if (accumulate) out[row] += s; else out[row] = s;
  }
}

// ---------------- attention scores: a[b][l] = sum_j wattn[b][j]*tanh((Wy[b]@Y[b])[j][l] + whn[b][j]) ----------------
// Y[b][h][l] = hspre[l][b][h]. Batched WMMA GEMM, Wy converted f32->f16 on the fly.
__global__ __launch_bounds__(256) void attn_score(const float* __restrict__ Wy,    // [B][H][H]
                                                  const h16* __restrict__ hspre,   // [L][B][H]
                                                  const float* __restrict__ whn,   // [B][H]
                                                  const float* __restrict__ wattn, // [B][H]
                                                  float* __restrict__ ascore) {    // [B][L]
  const int b = blockIdx.y;        // batch
  const int jc = blockIdx.x;       // j-chunk 0..7 (128 rows each)
  const int tid = threadIdx.x;
  const int lane = tid & 31;
  const int w = tid >> 5;
  const int j0 = jc * 128 + w * 16;          // this wave's 16 rows of M
  const float* A = Wy + (size_t)b * H_ * H_;
  const h16* Bb = hspre + (size_t)b * H_;    // row l lives at Bb + l*(B_*H_)

  v8f acc[8] = {};
  for (int k = 0; k < H_; k += 32) {
    v16h a = load_a_f32(A, H_, j0, k, lane);
#pragma unroll
    for (int nt = 0; nt < 8; ++nt) {
      v16h bm = load_b_f16(Bb, B_ * H_, nt * 16, k, lane);
      acc[nt] = wmma16(a, bm, acc[nt]);
    }
  }
  const int n = lane & 15;
  const int g = lane >> 4;
#pragma unroll
  for (int nt = 0; nt < 8; ++nt) {
    float s = 0.f;
#pragma unroll
    for (int r0 = 0; r0 < 8; ++r0) {
      int j = j0 + r0 + 8 * g;
      float m = tanhf(acc[nt][r0] + whn[(size_t)b * H_ + j]);
      s += wattn[(size_t)b * H_ + j] * m;
    }
    atomicAdd(&ascore[b * LPRE + nt * 16 + n], s);
  }
}

// ---------------- softmax over L=128 per batch (in place) ----------------
__global__ void softmax_l(float* __restrict__ a) {
  __shared__ float red[128];
  int b = blockIdx.x;
  int t = threadIdx.x;  // 128 threads
  float v = a[b * LPRE + t];
  red[t] = v;
  __syncthreads();
  for (int o = 64; o; o >>= 1) { if (t < o) red[t] = fmaxf(red[t], red[t + o]); __syncthreads(); }
  float m = red[0];
  __syncthreads();
  float e = __expf(v - m);
  red[t] = e;
  __syncthreads();
  for (int o = 64; o; o >>= 1) { if (t < o) red[t] += red[t + o]; __syncthreads(); }
  a[b * LPRE + t] = e / red[0];
}

// ---------------- r[b][h] = sum_l alpha[b][l] * hspre[l][b][h] ----------------
__global__ void compute_r(const h16* __restrict__ hspre, const float* __restrict__ alpha,
                          float* __restrict__ r) {
  int id = blockIdx.x * blockDim.x + threadIdx.x;  // 32768
  int b = id >> 10;
  int h = id & 1023;
  float s = 0.f;
  for (int l = 0; l < LPRE; ++l)
    s += alpha[b * LPRE + l] * (float)hspre[((size_t)l * B_ + b) * H_ + h];
  r[id] = s;
}

// ---------------- classifier + NLL loss (tiny: 32x3 logits) ----------------
__global__ void classify_loss(const float* __restrict__ t1,   // [B][H] pre-tanh h_star
                              const float* __restrict__ clsW, // [NCLS][H]
                              const float* __restrict__ clsb, // [NCLS]
                              const int* __restrict__ target,
                              float* __restrict__ out) {
  __shared__ float hs[H_];
  __shared__ float red[256];
  __shared__ float logits[NCLS_];
  int t = threadIdx.x;
  float loss = 0.f;
  for (int b = 0; b < B_; ++b) {
    for (int h = t; h < H_; h += 256) hs[h] = tanhf(t1[b * H_ + h]);
    __syncthreads();
    for (int c = 0; c < NCLS_; ++c) {
      float s = 0.f;
      for (int h = t; h < H_; h += 256) s += hs[h] * clsW[c * H_ + h];
      red[t] = s;
      __syncthreads();
      for (int o = 128; o; o >>= 1) { if (t < o) red[t] += red[t + o]; __syncthreads(); }
      if (t == 0) logits[c] = red[0] + clsb[c];
      __syncthreads();
    }
    if (t == 0) {
      float m = fmaxf(logits[0], fmaxf(logits[1], logits[2]));
      float lse = m + logf(__expf(logits[0] - m) + __expf(logits[1] - m) + __expf(logits[2] - m));
      loss += lse - logits[target[b]];
    }
    __syncthreads();
  }
  if (t == 0) out[0] = loss / (float)B_;
}

// ---------------- launch ----------------
extern "C" void kernel_launch(void* const* d_in, const int* in_sizes, int n_in,
                              void* d_out, int out_size, void* d_ws, size_t ws_size,
                              hipStream_t stream) {
  (void)in_sizes; (void)n_in; (void)out_size; (void)ws_size;
  const int*   X_pre  = (const int*)  d_in[0];
  const int*   X_hyp  = (const int*)  d_in[1];
  const int*   target = (const int*)  d_in[2];
  const float* emb    = (const float*)d_in[3];
  const float* W_ih_A = (const float*)d_in[4];
  const float* W_hh_A = (const float*)d_in[5];
  const float* b_ih_A = (const float*)d_in[6];
  const float* b_hh_A = (const float*)d_in[7];
  const float* W_ih_B = (const float*)d_in[8];
  const float* W_hh_B = (const float*)d_in[9];
  const float* b_ih_B = (const float*)d_in[10];
  const float* b_hh_B = (const float*)d_in[11];
  const float* W_y    = (const float*)d_in[12];
  const float* W_h    = (const float*)d_in[13];
  const float* w_attn = (const float*)d_in[14];
  const float* W_p    = (const float*)d_in[15];
  const float* W_x    = (const float*)d_in[16];
  const float* cls_W  = (const float*)d_in[17];
  const float* cls_b  = (const float*)d_in[18];

  char* ws = (char*)d_ws;
  size_t off = 0;
  auto alloc = [&](size_t bytes) -> char* {
    char* p = ws + off;
    off = (off + bytes + 255) & ~(size_t)255;
    return p;
  };
  h16*   wihA16 = (h16*)  alloc((size_t)G4H * E_ * 2);
  h16*   whhA16 = (h16*)  alloc((size_t)G4H * H_ * 2);
  h16*   wihB16 = (h16*)  alloc((size_t)G4H * E_ * 2);
  h16*   whhB16 = (h16*)  alloc((size_t)G4H * H_ * 2);
  h16*   xp     = (h16*)  alloc((size_t)LPRE * B_ * E_ * 2);
  h16*   xh     = (h16*)  alloc((size_t)LHYP * B_ * E_ * 2);
  h16*   hspre  = (h16*)  alloc((size_t)LPRE * B_ * H_ * 2);
  float* biasA  = (float*)alloc((size_t)G4H * 4);
  float* biasB  = (float*)alloc((size_t)G4H * 4);
  h16*   hbuf   = (h16*)  alloc((size_t)2 * B_ * H_ * 2);
  float* hf32   = (float*)alloc((size_t)B_ * H_ * 4);
  float* whn    = (float*)alloc((size_t)B_ * H_ * 4);
  float* ascore = (float*)alloc((size_t)B_ * LPRE * 4);
  float* rvec   = (float*)alloc((size_t)B_ * H_ * 4);
  float* t1     = (float*)alloc((size_t)B_ * H_ * 4);
  int*   ctr    = (int*)  alloc(256);

  prep_kernel<<<1024, 256, 0, stream>>>(emb, X_pre, X_hyp,
                                        W_ih_A, W_hh_A, b_ih_A, b_hh_A,
                                        W_ih_B, W_hh_B, b_ih_B, b_hh_B,
                                        wihA16, whhA16, wihB16, whhB16,
                                        xp, xh, biasA, biasB, hbuf, ascore, ctr);

  lstm_persistent<<<32, 256, LDS_BYTES, stream>>>(xp, xh, wihA16, whhA16, wihB16, whhB16,
                                                  biasA, biasB, hbuf, hf32, hspre, ctr);

  // whn = W_h @ h_n
  gemv_bhh<<<4096, 256, 0, stream>>>(W_h, hf32, whn, 0);
  // attention scores with WMMA batched GEMM
  attn_score<<<dim3(8, B_), 256, 0, stream>>>(W_y, hspre, whn, w_attn, ascore);
  softmax_l<<<B_, 128, 0, stream>>>(ascore);
  compute_r<<<128, 256, 0, stream>>>(hspre, ascore, rvec);
  // t1 = W_p @ r + W_x @ h_n   (tanh applied in classifier)
  gemv_bhh<<<4096, 256, 0, stream>>>(W_p, rvec, t1, 0);
  gemv_bhh<<<4096, 256, 0, stream>>>(W_x, hf32, t1, 1);
  classify_loss<<<1, 256, 0, stream>>>(t1, cls_W, cls_b, target, (float*)d_out);
}
